// GPT_61272003445330
// MI455X (gfx1250) — matrix operand order repaired and checked by reading, not compiled, not run on hardware
//
#include <hip/hip_runtime.h>
#include <hip/hip_bf16.h>

// ---------------- model constants ----------------
#define VOCAB  32000
#define SEQ    512
#define BATCH  8
#define EDIM   768
#define NHEAD  12
#define HDIM   64
#define NLAYER 6
#define FDIM   2048
#define TOK    (BATCH * SEQ)          // 4096 tokens
#define EPSF   1.1920929e-07f

typedef __attribute__((ext_vector_type(16))) _Float16 v16h;
typedef __attribute__((ext_vector_type(8)))  _Float16 v8h;
typedef __attribute__((ext_vector_type(4)))  _Float16 v4h;
typedef __attribute__((ext_vector_type(8)))  float    v8f;
typedef __attribute__((ext_vector_type(4)))  float    f32x4;

// =====================================================================
// WMMA GEMM:  C[M,N] (+= R) = A[M,K] * B
//   TRANS_B = false : B row-major [K,N] (weights, V)
//   TRANS_B = true  : B row-major [N,K] (K-matrix for Q*K^T)
// Block tile 128x64, K-step 32, 256 threads = 8 wave32s (4x2 wave grid),
// each wave computes 32x32 via 2x2 v_wmma_f32_16x16x32_f16.
// LDS double-buffered: next tile's global_load_b128s are issued before the
// current tile's ds_load_b128 fragments + WMMAs, then converted/stored into
// the alternate buffer. One barrier per k-step.
// =====================================================================
#define BM 128
#define BN 64
#define BK 32
#define LDPAD 40   // halves per LDS row (keeps fragment ds_loads 16B-aligned)

template<bool TRANS_B, bool RESID>
__global__ __launch_bounds__(256)
void gemm_wmma(const float* __restrict__ A, const float* __restrict__ B,
               float* __restrict__ C, const float* __restrict__ R,
               int M, int N, int K, int lda, int ldb, int ldc,
               long long sA, long long sB, long long sC)
{
    __shared__ __align__(16) _Float16 lA[2][BM * LDPAD];
    __shared__ __align__(16) _Float16 lB[2][BN * LDPAD];

    const int t    = threadIdx.x;
    const int lane = t & 31;
    const int w    = t >> 5;
    const int wm   = (w >> 1) * 32;     // wave row offset in tile
    const int wn   = (w & 1) * 32;      // wave col offset in tile
    const int m0   = blockIdx.y * BM;
    const int n0   = blockIdx.x * BN;

    A += (long long)blockIdx.z * sA;
    B += (long long)blockIdx.z * sB;
    C += (long long)blockIdx.z * sC;
    if (RESID) R += (long long)blockIdx.z * sC;

    v8f acc[2][2] = {};
    f32x4 ra[4];        // staged A regs: 4 x float4 per thread (128x32 tile)
    f32x4 rb[2];        // staged B regs: 2 x float4 per thread (64x32 tile)

    // ---- issue global_load_b128s for one k-tile into registers ----
    auto loadG = [&](int kk) {
#pragma unroll
        for (int i = 0; i < 4; ++i) {
            int idx = t + i * 256;                 // 0..1023
            int m = idx >> 3, k4 = (idx & 7) * 4;  // coalesced along k
            ra[i] = *(const f32x4*)(A + (size_t)(m0 + m) * lda + kk + k4);
        }
#pragma unroll
        for (int i = 0; i < 2; ++i) {
            int idx = t + i * 256;                 // 0..511
            if (TRANS_B) {
                int n = idx >> 3, k4 = (idx & 7) * 4;
                rb[i] = *(const f32x4*)(B + (size_t)(n0 + n) * ldb + kk + k4);
            } else {
                int kq = idx >> 4, n4 = (idx & 15) * 4;
                rb[i] = *(const f32x4*)(B + (size_t)(kk + kq) * ldb + n0 + n4);
            }
        }
    };

    // ---- convert f32->f16 (packed v_cvt_pk_rtz) and store into LDS buffer ----
    auto storeL = [&](int buf) {
#pragma unroll
        for (int i = 0; i < 4; ++i) {
            int idx = t + i * 256;
            int m = idx >> 3, k4 = (idx & 7) * 4;
            auto lo = __builtin_amdgcn_cvt_pkrtz(ra[i][0], ra[i][1]);
            auto hi = __builtin_amdgcn_cvt_pkrtz(ra[i][2], ra[i][3]);
            v4h h4;
            h4[0] = (_Float16)lo[0]; h4[1] = (_Float16)lo[1];
            h4[2] = (_Float16)hi[0]; h4[3] = (_Float16)hi[1];
            *(v4h*)&lA[buf][m * LDPAD + k4] = h4;
        }
#pragma unroll
        for (int i = 0; i < 2; ++i) {
            int idx = t + i * 256;
            if (TRANS_B) {
                int n = idx >> 3, k4 = (idx & 7) * 4;
                auto lo = __builtin_amdgcn_cvt_pkrtz(rb[i][0], rb[i][1]);
                auto hi = __builtin_amdgcn_cvt_pkrtz(rb[i][2], rb[i][3]);
                v4h h4;
                h4[0] = (_Float16)lo[0]; h4[1] = (_Float16)lo[1];
                h4[2] = (_Float16)hi[0]; h4[3] = (_Float16)hi[1];
                *(v4h*)&lB[buf][n * LDPAD + k4] = h4;
            } else {                               // transpose [k][n] -> [n][k]
                int kq = idx >> 4, n4 = (idx & 15) * 4;
#pragma unroll
                for (int j = 0; j < 4; ++j)
                    lB[buf][(n4 + j) * LDPAD + kq] = (_Float16)rb[i][j];
            }
        }
    };

    // ---- fragment loads (ISA 7.12.2 layouts) + 4 WMMAs ----
    auto compute = [&](int buf) {
        const int klo = (lane >> 4) * 8;   // A lane-group K split {0..7}/{8..15}
        v16h af[2], bfr[2];
#pragma unroll
        for (int im = 0; im < 2; ++im) {
            int row = wm + im * 16 + (lane & 15);
            v8h lo = *(const v8h*)&lA[buf][row * LDPAD + klo];
            v8h hi = *(const v8h*)&lA[buf][row * LDPAD + klo + 16];
#pragma unroll
            for (int j = 0; j < 8; ++j) { af[im][j] = lo[j]; af[im][j + 8] = hi[j]; }
        }
        const int kb = (lane >> 4) * 8;    // B lane-group K split {0..7}/{8..15}, same as A
#pragma unroll
        for (int in = 0; in < 2; ++in) {
            int col = wn + in * 16 + (lane & 15);
            v8h lo = *(const v8h*)&lB[buf][col * LDPAD + kb];
            v8h hi = *(const v8h*)&lB[buf][col * LDPAD + kb + 16];
#pragma unroll
            for (int j = 0; j < 8; ++j) { bfr[in][j] = lo[j]; bfr[in][j + 8] = hi[j]; }
        }
#pragma unroll
        for (int im = 0; im < 2; ++im)
#pragma unroll
            for (int in = 0; in < 2; ++in)
                acc[im][in] = __builtin_amdgcn_wmma_f32_16x16x32_f16(
                    false, af[im], false, bfr[in], (short)0, acc[im][in], false, false);
    };

    // ---- software-pipelined main loop ----
    const int nk = K / BK;
    loadG(0);
    storeL(0);
    int cur = 0;
    for (int kt = 0; kt < nk; ++kt) {
        __syncthreads();                       // buf[cur] visible to all waves
        if (kt + 1 < nk) loadG((kt + 1) * BK); // overlap next global tile
        compute(cur);
        if (kt + 1 < nk) { storeL(cur ^ 1); cur ^= 1; }
    }

    // ---- store: VGPR r -> rows r / r+8, N = lane&15 ----
    const int rofs = (lane >> 4) * 8;
    const int colL = lane & 15;
#pragma unroll
    for (int im = 0; im < 2; ++im)
#pragma unroll
        for (int in = 0; in < 2; ++in)
#pragma unroll
            for (int r = 0; r < 8; ++r) {
                int row = m0 + wm + im * 16 + r + rofs;
                int col = n0 + wn + in * 16 + colL;
                size_t o = (size_t)row * ldc + col;
                float vv = acc[im][in][r];
                if (RESID) vv += R[o];
                C[o] = vv;
            }
}

// =====================================================================
// Glue kernels (VALU)
// =====================================================================
__global__ void embed_gather(const int* __restrict__ ids,
                             const float* __restrict__ emb,
                             float* __restrict__ x)
{
    int idx = blockIdx.x * 256 + threadIdx.x;      // TOK*EDIM
    int tok = idx / EDIM, e = idx - tok * EDIM;
    x[idx] = emb[(size_t)ids[tok] * EDIM + e];
}

// one wave per row of EDIM=768 (24 elems/lane)
__global__ void rmsnorm_rows(const float* __restrict__ x,
                             const float* __restrict__ wgt,
                             float* __restrict__ out)
{
    int lane = threadIdx.x & 31;
    int row  = blockIdx.x * 8 + (threadIdx.x >> 5);
    const float* px = x + (size_t)row * EDIM;
    float vals[24]; float ss = 0.f;
#pragma unroll
    for (int j = 0; j < 24; ++j) { float v = px[j * 32 + lane]; vals[j] = v; ss += v * v; }
#pragma unroll
    for (int off = 16; off; off >>= 1) ss += __shfl_xor(ss, off, 32);
    float rs = rsqrtf(ss * (1.0f / EDIM) + EPSF);
    float* po = out + (size_t)row * EDIM;
#pragma unroll
    for (int j = 0; j < 24; ++j) po[j * 32 + lane] = vals[j] * rs * wgt[j * 32 + lane];
}

// in-place RoPE on q and k, layout [token, h*64+d]
__global__ void rope_qk(float* __restrict__ q, float* __restrict__ k)
{
    int idx = blockIdx.x * 256 + threadIdx.x;      // TOK*NHEAD*32
    int i    = idx & 31;
    int rest = idx >> 5;
    int h    = rest % NHEAD;
    int tok  = rest / NHEAD;
    int pos  = tok & (SEQ - 1);
    float ang = (float)pos * __powf(10000.0f, -(float)i * (1.0f / 32.0f));
    float s, c; __sincosf(ang, &s, &c);
    size_t base = (size_t)tok * EDIM + h * HDIM + 2 * i;
    float qe = q[base], qo = q[base + 1];
    q[base] = qe * c - qo * s;  q[base + 1] = qe * s + qo * c;
    float ke = k[base], ko = k[base + 1];
    k[base] = ke * c - ko * s;  k[base + 1] = ke * s + ko * c;
}

// causal softmax over scores[NHEAD, SEQ, SEQ] for one sequence; scale 1/sqrt(64)
__global__ void softmax_causal(float* __restrict__ sc)
{
    int lane = threadIdx.x & 31;
    int rowg = blockIdx.x * 8 + (threadIdx.x >> 5);   // 0 .. NHEAD*SEQ-1
    int qpos = rowg & (SEQ - 1);
    float* p = sc + (size_t)rowg * SEQ;
    float vals[16]; float mx = -INFINITY;
#pragma unroll
    for (int j = 0; j < 16; ++j) {
        int col = j * 32 + lane;
        float v = p[col] * 0.125f;
        v = (col <= qpos) ? v : -INFINITY;
        vals[j] = v; mx = fmaxf(mx, v);
    }
#pragma unroll
    for (int off = 16; off; off >>= 1) mx = fmaxf(mx, __shfl_xor(mx, off, 32));
    float sum = 0.f;
#pragma unroll
    for (int j = 0; j < 16; ++j) { vals[j] = __expf(vals[j] - mx); sum += vals[j]; }
#pragma unroll
    for (int off = 16; off; off >>= 1) sum += __shfl_xor(sum, off, 32);
    float inv = 1.0f / sum;
#pragma unroll
    for (int j = 0; j < 16; ++j) p[j * 32 + lane] = vals[j] * inv;
}

// gate = silu(gate) * val
__global__ void silu_gate(float* __restrict__ g, const float* __restrict__ v)
{
    int idx = blockIdx.x * 256 + threadIdx.x;      // TOK*FDIM
    float gv = g[idx];
    g[idx] = gv * (1.0f / (1.0f + __expf(-gv))) * v[idx];
}

// =====================================================================
// Host-side orchestration
// =====================================================================
static inline void launch_gemm_nn(const float* A, const float* B, float* Cp, const float* R,
                                  int M, int N, int K, int lda, int ldb, int ldc,
                                  long long sA, long long sB, long long sC, int batch,
                                  bool resid, hipStream_t st)
{
    dim3 grid(N / BN, M / BM, batch);
    if (resid)
        gemm_wmma<false, true ><<<grid, 256, 0, st>>>(A, B, Cp, R, M, N, K, lda, ldb, ldc, sA, sB, sC);
    else
        gemm_wmma<false, false><<<grid, 256, 0, st>>>(A, B, Cp, nullptr, M, N, K, lda, ldb, ldc, sA, sB, sC);
}

extern "C" void kernel_launch(void* const* d_in, const int* in_sizes, int n_in,
                              void* d_out, int out_size, void* d_ws, size_t ws_size,
                              hipStream_t stream)
{
    const int*   ids     = (const int*)  d_in[0];
    const float* embed   = (const float*)d_in[1];
    const float* norm1_w = (const float*)d_in[2];
    const float* wq      = (const float*)d_in[3];
    const float* wk      = (const float*)d_in[4];
    const float* wv      = (const float*)d_in[5];
    const float* wo      = (const float*)d_in[6];
    const float* norm2_w = (const float*)d_in[7];
    const float* w_gate  = (const float*)d_in[8];
    const float* w_val   = (const float*)d_in[9];
    const float* w_out   = (const float*)d_in[10];
    const float* normf_w = (const float*)d_in[11];
    const float* unembed = (const float*)d_in[12];
    float* out = (float*)d_out;

    // workspace layout (floats)
    float* ws   = (float*)d_ws;
    float* x    = ws;                                  // TOK*EDIM
    float* h    = x    + (size_t)TOK * EDIM;
    float* q    = h    + (size_t)TOK * EDIM;
    float* kbuf = q    + (size_t)TOK * EDIM;
    float* v    = kbuf + (size_t)TOK * EDIM;
    float* attn = v    + (size_t)TOK * EDIM;
    float* sc   = attn + (size_t)TOK * EDIM;           // NHEAD*SEQ*SEQ (per-sequence)
    float* gate = sc   + (size_t)NHEAD * SEQ * SEQ;    // TOK*FDIM
    float* valb = gate + (size_t)TOK * FDIM;           // TOK*FDIM

    const long long HC2 = (long long)SEQ * SEQ;

    // x = embed[ids]
    embed_gather<<<(TOK * EDIM) / 256, 256, 0, stream>>>(ids, embed, x);

    for (int l = 0; l < NLAYER; ++l) {
        // h = rmsnorm(x, norm1_w[l])
        rmsnorm_rows<<<TOK / 8, 256, 0, stream>>>(x, norm1_w + (size_t)l * EDIM, h);

        // q,k,v projections
        launch_gemm_nn(h, wq + (size_t)l * EDIM * EDIM, q,    nullptr, TOK, EDIM, EDIM, EDIM, EDIM, EDIM, 0, 0, 0, 1, false, stream);
        launch_gemm_nn(h, wk + (size_t)l * EDIM * EDIM, kbuf, nullptr, TOK, EDIM, EDIM, EDIM, EDIM, EDIM, 0, 0, 0, 1, false, stream);
        launch_gemm_nn(h, wv + (size_t)l * EDIM * EDIM, v,    nullptr, TOK, EDIM, EDIM, EDIM, EDIM, EDIM, 0, 0, 0, 1, false, stream);

        // RoPE
        rope_qk<<<(TOK * NHEAD * 32) / 256, 256, 0, stream>>>(q, kbuf);

        // attention, per sequence b; batch over heads in grid.z
        for (int b = 0; b < BATCH; ++b) {
            const float* qb = q    + (size_t)b * SEQ * EDIM;
            const float* kb = kbuf + (size_t)b * SEQ * EDIM;
            const float* vb = v    + (size_t)b * SEQ * EDIM;
            float*       ab = attn + (size_t)b * SEQ * EDIM;
            // scores = Q * K^T   (A strided per head by 64; B trans)
            {
                dim3 grid(SEQ / BN, SEQ / BM, NHEAD);
                gemm_wmma<true, false><<<grid, 256, 0, stream>>>(
                    qb, kb, sc, nullptr, SEQ, SEQ, HDIM,
                    EDIM, EDIM, SEQ, (long long)HDIM, (long long)HDIM, HC2);
            }
            // causal softmax (scale folded in)
            softmax_causal<<<(NHEAD * SEQ) / 8, 256, 0, stream>>>(sc);
            // attn = P * V
            {
                dim3 grid(HDIM / BN, SEQ / BM, NHEAD);
                gemm_wmma<false, false><<<grid, 256, 0, stream>>>(
                    sc, vb, ab, nullptr, SEQ, HDIM, SEQ,
                    SEQ, EDIM, EDIM, HC2, (long long)HDIM, (long long)HDIM);
            }
        }

        // x = x + attn @ wo[l]
        launch_gemm_nn(attn, wo + (size_t)l * EDIM * EDIM, x, x, TOK, EDIM, EDIM, EDIM, EDIM, EDIM, 0, 0, 0, 1, true, stream);

        // h = rmsnorm(x, norm2_w[l]) ; FFN
        rmsnorm_rows<<<TOK / 8, 256, 0, stream>>>(x, norm2_w + (size_t)l * EDIM, h);
        launch_gemm_nn(h, w_gate + (size_t)l * EDIM * FDIM, gate, nullptr, TOK, FDIM, EDIM, EDIM, FDIM, FDIM, 0, 0, 0, 1, false, stream);
        launch_gemm_nn(h, w_val  + (size_t)l * EDIM * FDIM, valb, nullptr, TOK, FDIM, EDIM, EDIM, FDIM, FDIM, 0, 0, 0, 1, false, stream);
        silu_gate<<<(TOK * FDIM) / 256, 256, 0, stream>>>(gate, valb);
        // x = x + gated @ w_out[l]
        launch_gemm_nn(gate, w_out + (size_t)l * FDIM * EDIM, x, x, TOK, EDIM, FDIM, FDIM, EDIM, EDIM, 0, 0, 0, 1, true, stream);
    }

    // final norm + unembed straight into d_out
    rmsnorm_rows<<<TOK / 8, 256, 0, stream>>>(x, normf_w, h);
    launch_gemm_nn(h, unembed, out, nullptr, TOK, VOCAB, EDIM, EDIM, VOCAB, VOCAB, 0, 0, 0, 1, false, stream);
}
